// LSTM0_83872121357050
// MI455X (gfx1250) — matrix operand order repaired and checked by reading, not compiled, not run on hardware
//
#include <hip/hip_runtime.h>

typedef _Float16 v16h __attribute__((ext_vector_type(16)));
typedef _Float16 v8h  __attribute__((ext_vector_type(8)));
typedef float    v8f  __attribute__((ext_vector_type(8)));
typedef float    v4f  __attribute__((ext_vector_type(4)));

union F16x16 { v16h v; v8h h[2]; _Float16 e[16]; };
union F32x8  { v8f v; v4f q[2]; float e[8]; };

// Problem sizes
constexpr int T_   = 512;
constexpr int NT_FC = 26;          // ceil(402/16)

// Workspace layout (bytes)
constexpr size_t OFF_WHH  = 0;                         // 8*8*8*32*16 f16 = 512 KB
constexpr size_t OFF_WIH  = 524288;                    // 8*8*2*32*16 f16 = 128 KB
constexpr size_t OFF_WFC  = OFF_WIH + 131072;          // 26*8*32*16 f16 = 208 KB
constexpr size_t OFF_BSUM = OFF_WFC + 212992;          // 1024 f32
constexpr size_t OFF_XG   = OFF_BSUM + 4096;           // 4*512*8*8*32*8 f32 = 128 MB
constexpr size_t OFF_HS   = OFF_XG + 134217728ull;     // 4*512*8*512 f16 = 16 MB

__device__ __forceinline__ float sigf(float x)   { return 1.f / (1.f + __expf(-x)); }
__device__ __forceinline__ float tanhf_(float x) { return 1.f - 2.f / (__expf(2.f * x) + 1.f); }

// ---------------------------------------------------------------------------
// Pack weights into WMMA B-fragment layout (lane = N column, 16 contiguous K
// halves per lane), f32 -> f16, and pre-sum biases.
// wave-task q: [0,512) whh   : (w,tl,kt) = (q/64, (q/8)%8, q%8)
//             [512,640) wih  : (w,tl,kt) = (q2/16, (q2/2)%8, q2%2)
//             [640,848) wfc  : (nt,kt)   = (q3/8, q3%8)
//             [848,880) bsum
// ---------------------------------------------------------------------------
__global__ __launch_bounds__(256) void pack_kernel(
    const float* __restrict__ W_ih, const float* __restrict__ W_hh,
    const float* __restrict__ b_ih, const float* __restrict__ b_hh,
    const float* __restrict__ W_fc, char* __restrict__ ws) {
  _Float16* whh  = (_Float16*)(ws + OFF_WHH);
  _Float16* wih  = (_Float16*)(ws + OFF_WIH);
  _Float16* wfc  = (_Float16*)(ws + OFF_WFC);
  float*    bsum = (float*)(ws + OFF_BSUM);

  const int l = threadIdx.x & 31;
  const int q = (blockIdx.x * blockDim.x + threadIdx.x) >> 5;

  if (q < 512) {                                   // W_hh [1024,256]
    const int w = q >> 6, tl = (q >> 3) & 7, kt = q & 7;
    const int g = tl >> 1, ht = tl & 1;
    const int row = g * 256 + w * 32 + ht * 16 + (l & 15);
    const int k0  = kt * 32 + (l >> 4) * 16;
    _Float16* dst = whh + ((size_t)q * 32 + l) * 16;
#pragma unroll
    for (int j = 0; j < 16; ++j) dst[j] = (_Float16)W_hh[row * 256 + k0 + j];
  } else if (q < 640) {                            // W_ih [1024,60] pad K->64
    const int q2 = q - 512;
    const int w = q2 >> 4, tl = (q2 >> 1) & 7, kt = q2 & 1;
    const int g = tl >> 1, ht = tl & 1;
    const int row = g * 256 + w * 32 + ht * 16 + (l & 15);
    const int k0  = kt * 32 + (l >> 4) * 16;
    _Float16* dst = wih + ((size_t)q2 * 32 + l) * 16;
#pragma unroll
    for (int j = 0; j < 16; ++j) {
      const int k = k0 + j;
      dst[j] = (k < 60) ? (_Float16)W_ih[row * 60 + k] : (_Float16)0.f;
    }
  } else if (q < 848) {                            // W_fc [402,256] pad N->416
    const int q3 = q - 640;
    const int nt = q3 >> 3, kt = q3 & 7;
    const int n  = nt * 16 + (l & 15);
    const int k0 = kt * 32 + (l >> 4) * 16;
    _Float16* dst = wfc + ((size_t)q3 * 32 + l) * 16;
#pragma unroll
    for (int j = 0; j < 16; ++j)
      dst[j] = (n < 402) ? (_Float16)W_fc[n * 256 + k0 + j] : (_Float16)0.f;
  } else if (q < 880) {                            // b_ih + b_hh
    const int i = (q - 848) * 32 + l;
    bsum[i] = b_ih[i] + b_hh[i];
  }
}

// ---------------------------------------------------------------------------
// xg = x @ W_ih^T + (b_ih + b_hh), written in per-tile C-fragment layout
// (8 contiguous f32 per lane) for direct accumulator init in the scan.
// One block per M-tile (wg,t); wave w computes its 8 gate tiles.
// ---------------------------------------------------------------------------
__global__ __launch_bounds__(256) void xg_kernel(const float* __restrict__ x,
                                                 char* __restrict__ ws) {
  const _Float16* wih  = (const _Float16*)(ws + OFF_WIH);
  const float*    bsum = (const float*)(ws + OFF_BSUM);
  float*          xg   = (float*)(ws + OFF_XG);

  const int mt = blockIdx.x;             // wg*512 + t
  const int wg = mt >> 9, t = mt & 511;
  const int l = threadIdx.x & 31, w = threadIdx.x >> 5;

  // Build A fragments (16x64 of x, K padded 60->64)
  const int brow = wg * 16 + (l & 15);
  const int koff = (l >> 4) * 16;
  F16x16 a[2];
#pragma unroll
  for (int kt = 0; kt < 2; ++kt) {
#pragma unroll
    for (int j = 0; j < 16; ++j) {
      const int k = kt * 32 + koff + j;
      a[kt].e[j] = (k < 60) ? (_Float16)x[((size_t)brow * T_ + t) * 60 + k]
                            : (_Float16)0.f;
    }
  }

#pragma unroll
  for (int tl = 0; tl < 8; ++tl) {
    const int g = tl >> 1, ht = tl & 1;
    const int rowbase = g * 256 + w * 32 + ht * 16;
    const float bias = bsum[rowbase + (l & 15)];
    F32x8 acc;
#pragma unroll
    for (int e = 0; e < 8; ++e) acc.e[e] = bias;
#pragma unroll
    for (int kt = 0; kt < 2; ++kt) {
      const v8h* bp = (const v8h*)(wih + (((size_t)(w * 8 + tl) * 2 + kt) * 32 + l) * 16);
      F16x16 b; b.h[0] = bp[0]; b.h[1] = bp[1];
      acc.v = __builtin_amdgcn_wmma_f32_16x16x32_f16(false, a[kt].v, false, b.v,
                                                     (short)0, acc.v, false, false);
    }
    v4f* dst = (v4f*)(xg + ((((size_t)mt * 8 + w) * 8 + tl) * 32 + l) * 8);
    dst[0] = acc.q[0]; dst[1] = acc.q[1];
  }
}

// ---------------------------------------------------------------------------
// Recurrent scan. 4 workgroups, each owns 16 batch rows (fully independent
// scans -> no inter-WG sync). Wave w owns hidden units [32w,32w+32):
// gate tiles tl = g*2+ht for its slice. c stays in registers; h round-trips
// through LDS, which also performs the C-layout -> A-layout shuffle and
// feeds the packed hs export for the FC kernel.
// ---------------------------------------------------------------------------
__global__ __launch_bounds__(256) void lstm_kernel(char* __restrict__ ws) {
  __shared__ _Float16 hbuf[16 * 264];              // padded stride: bank-friendly

  const _Float16* whh = (const _Float16*)(ws + OFF_WHH);
  const float*    xg  = (const float*)(ws + OFF_XG);
  _Float16*       hs  = (_Float16*)(ws + OFF_HS);

  const int l = threadIdx.x & 31, w = threadIdx.x >> 5;
  const int wg = blockIdx.x;
  const int row = l & 15, hi8 = l >> 4;

  for (int i = threadIdx.x; i < 16 * 264; i += 256) hbuf[i] = (_Float16)0.f;
  __syncthreads();

  F32x8 c[2];
#pragma unroll
  for (int ht = 0; ht < 2; ++ht)
#pragma unroll
    for (int e = 0; e < 8; ++e) c[ht].e[e] = 0.f;

#pragma unroll 1
  for (int t = 0; t < T_; ++t) {
    // A fragments of h_{t-1} from LDS (full K = 256)
    F16x16 a[8];
#pragma unroll
    for (int kt = 0; kt < 8; ++kt) {
      const int k0 = kt * 32 + hi8 * 16;
      a[kt].h[0] = *(const v8h*)&hbuf[row * 264 + k0];
      a[kt].h[1] = *(const v8h*)&hbuf[row * 264 + k0 + 8];
    }
    __syncthreads();                               // WAR: reads done before overwrite

    if (t + 1 < T_) {                              // warm L2/L0 for next step's xg
      __builtin_prefetch(xg + ((((size_t)(wg * 512 + t + 1) * 8 + w) * 8) * 32 + l) * 8, 0, 0);
    }

#pragma unroll
    for (int ht = 0; ht < 2; ++ht) {
      F32x8 acc[4];
#pragma unroll
      for (int g = 0; g < 4; ++g) {
        const int tl = g * 2 + ht;
        const v4f* xp = (const v4f*)(xg + ((((size_t)(wg * 512 + t) * 8 + w) * 8 + tl) * 32 + l) * 8);
        acc[g].q[0] = xp[0]; acc[g].q[1] = xp[1];  // C := xg (pre-tiled)
#pragma unroll
        for (int kt = 0; kt < 8; ++kt) {
          const v8h* bp = (const v8h*)(whh + (((size_t)(w * 8 + tl) * 8 + kt) * 32 + l) * 16);
          F16x16 b; b.h[0] = bp[0]; b.h[1] = bp[1];
          acc[g].v = __builtin_amdgcn_wmma_f32_16x16x32_f16(false, a[kt].v, false, b.v,
                                                            (short)0, acc[g].v, false, false);
        }
      }
      // gates: i, f, g, o  -> cell/hidden update (f32, in registers)
      float hh[8];
#pragma unroll
      for (int e = 0; e < 8; ++e) {
        const float iv = sigf(acc[0].e[e]);
        const float fv = sigf(acc[1].e[e]);
        const float gv = tanhf_(acc[2].e[e]);
        const float ov = sigf(acc[3].e[e]);
        const float cv = fv * c[ht].e[e] + iv * gv;
        c[ht].e[e] = cv;
        hh[e] = ov * tanhf_(cv);
      }
      const int colb = w * 32 + ht * 16 + (l & 15);
#pragma unroll
      for (int r = 0; r < 8; ++r)
        hbuf[(r + hi8 * 8) * 264 + colb] = (_Float16)hh[r];
    }
    __syncthreads();                               // RAW: h_t visible to all waves

    // export h_t K-tile (kt == w) as a packed A-fragment for the FC GEMM
    {
      const int k0 = w * 32 + hi8 * 16;
      const v8h lo = *(const v8h*)&hbuf[row * 264 + k0];
      const v8h hi = *(const v8h*)&hbuf[row * 264 + k0 + 8];
      v8h* dst = (v8h*)(hs + (((size_t)(wg * 512 + t) * 8 + w) * 512 + (size_t)l * 16));
      dst[0] = lo; dst[1] = hi;
    }
  }
}

// ---------------------------------------------------------------------------
// out = hs @ W_fc^T + b_fc   (M=32768, N=402->416, K=256), f32 out.
// ---------------------------------------------------------------------------
__global__ __launch_bounds__(256) void fc_kernel(const float* __restrict__ b_fc,
                                                 const char* __restrict__ ws,
                                                 float* __restrict__ out) {
  const _Float16* hs  = (const _Float16*)(ws + OFF_HS);
  const _Float16* wfc = (const _Float16*)(ws + OFF_WFC);

  const int mt = blockIdx.x;             // wg*512 + t
  const int wg = mt >> 9, t = mt & 511;
  const int l = threadIdx.x & 31, w = threadIdx.x >> 5;

  F16x16 a[8];
#pragma unroll
  for (int kt = 0; kt < 8; ++kt) {
    const v8h* ap = (const v8h*)(hs + ((size_t)mt * 8 + kt) * 512 + (size_t)l * 16);
    a[kt].h[0] = ap[0]; a[kt].h[1] = ap[1];
  }

  for (int nt = w; nt < NT_FC; nt += 8) {
    const int n = nt * 16 + (l & 15);
    const float bias = (n < 402) ? b_fc[n] : 0.f;
    F32x8 acc;
#pragma unroll
    for (int e = 0; e < 8; ++e) acc.e[e] = bias;
#pragma unroll
    for (int kt = 0; kt < 8; ++kt) {
      const v8h* bp = (const v8h*)(wfc + (((size_t)nt * 8 + kt) * 32 + l) * 16);
      F16x16 b; b.h[0] = bp[0]; b.h[1] = bp[1];
      acc.v = __builtin_amdgcn_wmma_f32_16x16x32_f16(false, a[kt].v, false, b.v,
                                                     (short)0, acc.v, false, false);
    }
    if (n < 402) {
#pragma unroll
      for (int r = 0; r < 8; ++r) {
        const int brow = wg * 16 + r + (l >> 4) * 8;
        out[((size_t)brow * T_ + t) * 402 + n] = acc.e[r];
      }
    }
  }
}

extern "C" void kernel_launch(void* const* d_in, const int* in_sizes, int n_in,
                              void* d_out, int out_size, void* d_ws, size_t ws_size,
                              hipStream_t stream) {
  const float* x    = (const float*)d_in[0];
  const float* W_ih = (const float*)d_in[1];
  const float* W_hh = (const float*)d_in[2];
  const float* b_ih = (const float*)d_in[3];
  const float* b_hh = (const float*)d_in[4];
  const float* W_fc = (const float*)d_in[5];
  const float* b_fc = (const float*)d_in[6];
  char* ws = (char*)d_ws;
  float* out = (float*)d_out;

  pack_kernel<<<128, 256, 0, stream>>>(W_ih, W_hh, b_ih, b_hh, W_fc, ws);
  xg_kernel<<<2048, 256, 0, stream>>>(x, ws);
  lstm_kernel<<<4, 256, 0, stream>>>(ws);
  fc_kernel<<<2048, 256, 0, stream>>>(b_fc, ws, out);
}
